// GAT_60266981097984
// MI455X (gfx1250) — compile-verified, matrix operand order checked by reading, not run on hardware
//
#include <hip/hip_runtime.h>
#include <math.h>

// ---------------------------------------------------------------------------
// CDNA5 (gfx1250, wave32) GAT forward.
//  - Dense FC projections via v_wmma_f32_16x16x32_f16 (f32->f16 fragments,
//    f32 accumulate). Register-blocked: each wave holds all B fragments for
//    K=128 (4 x v16h) and streams 4 row tiles of A over them -> 16 unrolled
//    WMMAs per wave, contiguous A loads, 4x less B traffic.
//  - Edge softmax + scatter via L2 atomics (192MB L2 holds all node state).
// ---------------------------------------------------------------------------

typedef __attribute__((ext_vector_type(16))) _Float16 v16h;
typedef __attribute__((ext_vector_type(8)))  float    v8f;

#define NEG_SLOPE 0.2f

// ------------------------- WMMA GEMM: C = A(f32) x B(f32) -------------------
// A: MxK row-major, B: KxN row-major, C: MxN row-major. K compile-time, %32==0.
// grid.x covers ceil(M/16/ROWS) row-tile groups; (grid.y * waves/block) covers
// N/16 col tiles. One wave: 1 col tile x ROWS row tiles.
template <int K, int ROWS>
__global__ void wmma_gemm_f32f16(const float* __restrict__ A,
                                 const float* __restrict__ B,
                                 float* __restrict__ C,
                                 int M, int Nld) {
  constexpr int KSTEPS = K / 32;
  const int wave    = threadIdx.x >> 5;
  const int lane    = threadIdx.x & 31;
  const int colTile = blockIdx.y * (blockDim.x >> 5) + wave;
  if (colTile * 16 >= Nld) return;                 // wave-uniform: EXEC stays full
  const int col0 = colTile * 16;
  const int g    = lane >> 4;                      // 0/1 half-wave group
  const int ml   = lane & 15;

  // ---- load ALL B fragments once; reused across ROWS row tiles ----
  // B fragment: lane holds column N=ml, K runs [k0+g*8,+8) and [k0+16+g*8,+8)
  const float* bcol = B + col0 + ml;
  v16h bf[KSTEPS];
#pragma unroll
  for (int ks = 0; ks < KSTEPS; ++ks) {
    const int k0 = ks * 32;
#pragma unroll
    for (int j = 0; j < 8; ++j) {
      bf[ks][j]     = (_Float16)bcol[(size_t)(k0 + g * 8 + j) * Nld];
      bf[ks][j + 8] = (_Float16)bcol[(size_t)(k0 + 16 + g * 8 + j) * Nld];
    }
  }

  const int totalRowTiles = M >> 4;
  const int rowTile0      = blockIdx.x * ROWS;
#pragma unroll
  for (int r = 0; r < ROWS; ++r) {
    const int rowTile = rowTile0 + r;
    if (rowTile < totalRowTiles) {                 // wave-uniform guard
      const int row0 = rowTile * 16;
      const float* arow = A + (size_t)(row0 + ml) * K;   // lane holds row M=ml
      v8f c = {};
#pragma unroll
      for (int ks = 0; ks < KSTEPS; ++ks) {
        const int k0 = ks * 32;
        v16h a;
#pragma unroll
        for (int j = 0; j < 8; ++j) {
          // two contiguous 8-float runs per lane (ISA 7.12.2, 16-bit A 16x32)
          a[j]     = (_Float16)arow[k0 + g * 8 + j];
          a[j + 8] = (_Float16)arow[k0 + 16 + g * 8 + j];
        }
        c = __builtin_amdgcn_wmma_f32_16x16x32_f16(false, a, false, bf[ks],
                                                   (short)0, c, false, false);
      }
      // D layout: VGPR rr -> row g*8+rr, col ml
      float* crow = C + (size_t)(row0 + g * 8) * Nld + col0 + ml;
#pragma unroll
      for (int rr = 0; rr < 8; ++rr) crow[(size_t)rr * Nld] = c[rr];
    }
  }
}

// ------------------------- small utility kernels ---------------------------
__global__ void fill_u32_k(unsigned* __restrict__ p, unsigned v, int n) {
  int i = blockIdx.x * blockDim.x + threadIdx.x;
  if (i < n) p[i] = v;
}

// order-preserving float<->uint encoding for atomicMax on floats
__device__ __forceinline__ unsigned enc_f(float f) {
  unsigned u = __float_as_uint(f);
  return (u & 0x80000000u) ? ~u : (u | 0x80000000u);
}
__device__ __forceinline__ float dec_f(unsigned e) {
  unsigned u = (e & 0x80000000u) ? (e & 0x7fffffffu) : ~e;
  return __uint_as_float(u);
}
#define ENC_NEG_INF 0x007FFFFFu   // enc_f(-inf)

// el/er: per-node per-head attention projections (out=32 fixed)
__global__ void scores_k(const float* __restrict__ feat,
                         const float* __restrict__ al,
                         const float* __restrict__ ar,
                         float* __restrict__ el, float* __restrict__ er,
                         int N, int H) {
  int idx = blockIdx.x * blockDim.x + threadIdx.x;
  if (idx >= N * H) return;
  int n = idx / H, h = idx % H;
  const float* f  = feat + (size_t)n * H * 32 + h * 32;
  const float* wl = al + h * 32;
  const float* wr = ar + h * 32;
  float sl = 0.f, sr = 0.f;
#pragma unroll
  for (int d = 0; d < 32; ++d) { float v = f[d]; sl += v * wl[d]; sr += v * wr[d]; }
  el[idx] = sl;
  er[idx] = sr;
}

// pass 1: per-edge leaky-relu score, store it, atomic-max into dst
__global__ void edge_score_max_k(const int* __restrict__ src,
                                 const int* __restrict__ dst,
                                 const float* __restrict__ el,
                                 const float* __restrict__ er,
                                 float* __restrict__ ex,
                                 unsigned* __restrict__ emax,
                                 int E, int H) {
  int idx = blockIdx.x * blockDim.x + threadIdx.x;
  if (idx >= E * H) return;
  int e = idx / H, h = idx % H;
  int s = src[e], d = dst[e];
  float v = el[s * H + h] + er[d * H + h];
  v = v > 0.f ? v : NEG_SLOPE * v;
  ex[idx] = v;
  atomicMax(&emax[d * H + h], enc_f(v));
}

// pass 2: exp(score - max), accumulate denominator
__global__ void edge_exp_k(const int* __restrict__ dst,
                           float* __restrict__ ex,
                           const unsigned* __restrict__ emax,
                           float* __restrict__ denom,
                           int E, int H) {
  int idx = blockIdx.x * blockDim.x + threadIdx.x;
  if (idx >= E * H) return;
  int e = idx / H, h = idx % H;
  int d = dst[e];
  float z = __expf(ex[idx] - dec_f(emax[d * H + h]));
  ex[idx] = z;
  atomicAdd(&denom[d * H + h], z);
}

// pass 3: one wave32 per edge, lane = output channel (out==32).
// Lane h computes alpha for head h once; broadcast with __shfl (wave32).
__global__ void message_k(const int* __restrict__ src,
                          const int* __restrict__ dst,
                          const float* __restrict__ feat,
                          const float* __restrict__ ex,
                          const float* __restrict__ denom,
                          float* __restrict__ rst,
                          int E, int H) {
  int gw   = (blockIdx.x * blockDim.x + threadIdx.x) >> 5;
  int lane = threadIdx.x & 31;
  if (gw >= E) return;
  int s = src[gw], d = dst[gw];
  int HO = H * 32;
  float myAlpha = 0.f;
  if (lane < H)
    myAlpha = ex[(size_t)gw * H + lane] / fmaxf(denom[d * H + lane], 1e-9f);
  for (int h = 0; h < H; ++h) {
    float a = __shfl(myAlpha, h, 32);
    atomicAdd(&rst[(size_t)d * HO + h * 32 + lane],
              feat[(size_t)s * HO + h * 32 + lane] * a);
  }
}

// finalize: rst = relu(rst + bias [+ res])  (in place)
__global__ void fin_bias_relu_k(float* __restrict__ rst,
                                const float* __restrict__ bias,
                                const float* __restrict__ res,
                                int N, int HO) {
  int idx = blockIdx.x * blockDim.x + threadIdx.x;
  if (idx >= N * HO) return;
  int i = idx % HO;
  float v = rst[idx] + bias[i] + (res ? res[idx] : 0.f);
  rst[idx] = v > 0.f ? v : 0.f;
}

// layer-3 finalize: out[n][d] = mean_h( rst + res + b )  (H=6, out=32, no relu)
__global__ void fin3_mean_k(const float* __restrict__ rst,
                            const float* __restrict__ res,
                            const float* __restrict__ bias,
                            float* __restrict__ out, int N) {
  int idx = blockIdx.x * blockDim.x + threadIdx.x;
  if (idx >= N * 32) return;
  int n = idx >> 5, d = idx & 31;
  float s = 0.f;
#pragma unroll
  for (int h = 0; h < 6; ++h) {
    size_t o = (size_t)n * 192 + h * 32 + d;
    s += rst[o] + res[o] + bias[h * 32 + d];
  }
  out[idx] = s * (1.f / 6.f);
}

// ---------------------------------------------------------------------------
static inline int cdiv(int a, int b) { return (a + b - 1) / b; }

static void attention_pass(const int* src, const int* dst, const float* feat,
                           const float* al, const float* ar,
                           float* el, float* er, unsigned* emax, float* denom,
                           float* ex, float* rst, int N, int E, int H,
                           hipStream_t stream) {
  const int T = 256;
  int HO = H * 32;
  fill_u32_k<<<cdiv(N * H, T), T, 0, stream>>>((unsigned*)denom, 0u, N * H);
  fill_u32_k<<<cdiv(N * H, T), T, 0, stream>>>(emax, ENC_NEG_INF, N * H);
  fill_u32_k<<<cdiv(N * HO, T), T, 0, stream>>>((unsigned*)rst, 0u, N * HO);
  scores_k<<<cdiv(N * H, T), T, 0, stream>>>(feat, al, ar, el, er, N, H);
  edge_score_max_k<<<cdiv(E * H, T), T, 0, stream>>>(src, dst, el, er, ex, emax, E, H);
  edge_exp_k<<<cdiv(E * H, T), T, 0, stream>>>(dst, ex, emax, denom, E, H);
  message_k<<<cdiv(E * 32, T), T, 0, stream>>>(src, dst, feat, ex, denom, rst, E, H);
}

extern "C" void kernel_launch(void* const* d_in, const int* in_sizes, int n_in,
                              void* d_out, int out_size, void* d_ws, size_t ws_size,
                              hipStream_t stream) {
  const float* x     = (const float*)d_in[0];
  const int*   src   = (const int*)  d_in[1];
  const int*   dst   = (const int*)  d_in[2];
  const float* W1    = (const float*)d_in[3];
  const float* al1   = (const float*)d_in[4];
  const float* ar1   = (const float*)d_in[5];
  const float* b1    = (const float*)d_in[6];
  const float* W2    = (const float*)d_in[7];
  const float* al2   = (const float*)d_in[8];
  const float* ar2   = (const float*)d_in[9];
  const float* b2    = (const float*)d_in[10];
  const float* W3    = (const float*)d_in[11];
  const float* al3   = (const float*)d_in[12];
  const float* ar3   = (const float*)d_in[13];
  const float* b3    = (const float*)d_in[14];
  const float* resW3 = (const float*)d_in[15];

  const int N = in_sizes[0] / 128;   // 50000 (multiple of 16: 3125 row tiles)
  const int E = in_sizes[1];         // 850000

  // ---- workspace carve (floats) ----
  float* ws = (float*)d_ws;
  const size_t NB = (size_t)N * 192;
  float*    B0    = ws;              // feat buffer (all layers)
  float*    B1    = B0 + NB;         // rst1/h1, later res3
  float*    B2    = B1 + NB;         // rst2/h2, later rst3
  float*    el    = B2 + NB;
  float*    er    = el + (size_t)N * 6;
  unsigned* emax  = (unsigned*)(er + (size_t)N * 6);
  float*    denom = (float*)(emax + (size_t)N * 6);
  float*    ex    = denom + (size_t)N * 6;

  const int T = 256;
  constexpr int ROWS = 4;
  const int rowTiles  = N / 16;                  // 3125
  const int rowGroups = cdiv(rowTiles, ROWS);    // 782
  dim3 blk(256);

  // ================= layer 1: x(128) -> 4 heads x 32 =================
  // feat1 = x @ W1   (K=128, Nout=128 -> 8 col tiles = 8 waves, grid.y=1)
  wmma_gemm_f32f16<128, ROWS><<<dim3(rowGroups, 1), blk, 0, stream>>>(x, W1, B0, N, 128);
  attention_pass(src, dst, B0, al1, ar1, el, er, emax, denom, ex, B1, N, E, 4, stream);
  fin_bias_relu_k<<<cdiv(N * 128, T), T, 0, stream>>>(B1, b1, nullptr, N, 128); // h1 in B1

  // ================= layer 2: h1(128) -> 4 heads x 32, identity residual ====
  wmma_gemm_f32f16<128, ROWS><<<dim3(rowGroups, 1), blk, 0, stream>>>(B1, W2, B0, N, 128);
  attention_pass(src, dst, B0, al2, ar2, el, er, emax, denom, ex, B2, N, E, 4, stream);
  fin_bias_relu_k<<<cdiv(N * 128, T), T, 0, stream>>>(B2, b2, B1, N, 128);      // h2 in B2

  // ================= layer 3: h2(128) -> 6 heads x 32, proj residual, mean ==
  // feat3 = h2 @ W3 ; res3 = h2 @ resW3   (Nout=192 -> 12 col tiles, grid.y=2)
  wmma_gemm_f32f16<128, ROWS><<<dim3(rowGroups, 2), blk, 0, stream>>>(B2, W3,    B0, N, 192);
  wmma_gemm_f32f16<128, ROWS><<<dim3(rowGroups, 2), blk, 0, stream>>>(B2, resW3, B1, N, 192);
  attention_pass(src, dst, B0, al3, ar3, el, er, emax, denom, ex, B2, N, E, 6, stream);
  fin3_mean_k<<<cdiv(N * 32, T), T, 0, stream>>>(B2, B1, b3, (float*)d_out, N);
}
